// GGNAT_61314953118003
// MI455X (gfx1250) — matrix operand order repaired and checked by reading, not compiled, not run on hardware
//
#include <hip/hip_runtime.h>
#include <hip/hip_bf16.h>
#include <cstdint>
#include <cstddef>

typedef __bf16 bf16_t;
typedef __attribute__((ext_vector_type(16))) __bf16 v16bf;
typedef __attribute__((ext_vector_type(8)))  float  v8f;

#define DEVINL static __device__ __forceinline__

// ---------------- problem constants ----------------
constexpr int   kB     = 16;
constexpr int   kNPG   = 2048;
constexpr int   kN     = kB * kNPG;   // 32768 nodes
constexpr int   kE     = 262144;      // edges
constexpr int   kD     = 128;
constexpr int   kH     = 3;
constexpr int   kNET   = 4;
constexpr int   kSTEPS = 8;
constexpr float kSLOPE = 0.2f;

// ---------------- workspace layout (bytes) ----------------
constexpr size_t MB1 = 1024ull * 1024ull;
constexpr size_t OFF_H    = 0;               // f32 h        16MB
constexpr size_t OFF_HB   = 16 * MB1;        // bf16 h        8MB
constexpr size_t RR       = 24 * MB1;        // reusable region
// phase 1 (GGNN)
constexpr size_t OFF_HALL = RR;              // bf16 NET*N*D 32MB
constexpr size_t OFF_A    = RR + 32 * MB1;   // f32 N*D      16MB
constexpr size_t OFF_ABF  = RR + 48 * MB1;   // bf16 N*D      8MB
constexpr size_t OFF_GI   = RR + 56 * MB1;   // f32 N*384    48MB
constexpr size_t OFF_GH   = RR + 104 * MB1;  // f32 N*384    48MB
// phase 2 (GAT) — aliases phase 1
constexpr size_t OFF_ZF   = RR;              // bf16 N*384   24MB
constexpr size_t OFF_EL   = RR + 48 * MB1;
constexpr size_t OFF_ER   = RR + 48 * MB1 + 512 * 1024;
constexpr size_t OFF_EMAX = RR + 49 * MB1;
constexpr size_t OFF_DEN  = RR + 49 * MB1 + 512 * 1024;
constexpr size_t OFF_EE   = RR + 50 * MB1;   // f32 E*H       3MB
constexpr size_t OFF_GOUT = RR + 56 * MB1;   // f32 N*384    48MB
// phase 3 (head) — aliases phase 2
constexpr size_t OFF_HIB  = RR;              // bf16 N*128    8MB
constexpr size_t OFF_CIB  = RR + 8 * MB1;    // bf16 N*256   16MB
constexpr size_t OFF_BUFA = RR + 24 * MB1;   // bf16 up to B*2046*256 ~16MB
constexpr size_t OFF_BUFB = RR + 40 * MB1;   // bf16 up to B*1022*256 ~8MB
constexpr size_t OFF_Y2   = RR + 48 * MB1;   // f32 B*511*128
constexpr size_t OFF_P2   = RR + 52 * MB1;
constexpr size_t OFF_Z2   = RR + 56 * MB1;   // f32 B*511*256
constexpr size_t OFF_U2   = RR + 64 * MB1;
constexpr size_t OFF_BA   = RR + 72 * MB1;   // f32 32
constexpr size_t OFF_WB   = 176 * MB1;       // bf16 weights ~1.7MB
// bf16-weight sub-offsets (elements)
constexpr size_t WB_WET = 0,      WB_WIH = 65536,  WB_WHH = 114688, WB_WG  = 163840,
                 WB_C1  = 212992, WB_A1  = 262144, WB_C2  = 311296, WB_A2  = 327680,
                 WB_CC1 = 344064, WB_AC1 = 540672, WB_CC2 = 737280, WB_AC2 = 802816;

// ---------------- helpers ----------------
DEVINL bf16_t f2bf(float f) {
  unsigned u = __float_as_uint(f);
  u += 0x7fffu + ((u >> 16) & 1u);          // round to nearest even
  unsigned short h = (unsigned short)(u >> 16);
  bf16_t b; __builtin_memcpy(&b, &h, 2); return b;
}

// A fragment: 16x32 bf16 tile, ISA 7.12.2 layout (row per lane, K pairs per VGPR)
// VGPR v<4: K = half+2v ; VGPR v>=4: K = 16+half+2(v-4)
DEVINL v16bf load_frag_A(const bf16_t* __restrict__ A, int lda, int row0, int kc,
                         int lane, int M) {
  int r = row0 + (lane & 15);
  if (r > M - 1) r = M - 1;                  // tail clamp (stores are guarded)
  const int half = (lane & 16) ? 8 : 0;
  const bf16_t* base = A + (long)r * lda + kc * 32;
  v16bf f;
#pragma unroll
  for (int v = 0; v < 4; ++v) {
    int k0 = half + 2 * v;
    f[2 * v]     = base[k0];
    f[2 * v + 1] = base[k0 + 1];
    int k1 = 16 + half + 2 * v;
    f[8 + 2 * v]     = base[k1];
    f[8 + 2 * v + 1] = base[k1 + 1];
  }
  return f;
}

// B fragment: 32x16, B[k][n] = W[n][k], W row-major (out,in): col per lane, K pairs per VGPR
DEVINL v16bf load_frag_B(const bf16_t* __restrict__ W, int ldw, int col0, int kc, int lane) {
  int n = col0 + (lane & 15);
  int kb = (lane & 16) ? 8 : 0;
  const bf16_t* base = W + (long)n * ldw + kc * 32 + kb;
  v16bf f;
#pragma unroll
  for (int v = 0; v < 8; ++v) { int ko = (v < 4) ? 2 * v : 16 + 2 * (v - 4); f[2 * v] = base[ko]; f[2 * v + 1] = base[ko + 1]; }
  return f;
}

DEVINL v8f wmma_bf16(v16bf a, v16bf b, v8f c) {
  return __builtin_amdgcn_wmma_f32_16x16x32_bf16(false, a, false, b, (short)0, c, false, false);
}

// ---------------- generic WMMA GEMM: C[bz] = A[bz] @ W[bz]^T (+bias) ----------------
// A: M x Kdim bf16 (row stride lda), W: Ncol x Kdim bf16, C: M x Ncol (f32 or bf16)
// One wave computes a 32x64 output tile (2 row-tiles x 4 col-tiles), A double-buffered.
template <bool OUTBF, bool RELU>
__global__ void k_wmma_gemm(const bf16_t* __restrict__ A, int lda, long aBatch,
                            const bf16_t* __restrict__ W, long wBatch,
                            const float* __restrict__ bias, long biasBatch,
                            void* __restrict__ Cout, long cBatch,
                            int M, int Ncol, int Kdim) {
  const int lane = threadIdx.x;                  // 0..31 (wave32)
  const int slab = blockIdx.x * 4 + threadIdx.y; // 4 waves / block
  const int colSlabs = Ncol >> 6;
  const int rowTiles = (M + 31) >> 5;            // 32-row tiles
  if (slab >= rowTiles * colSlabs) return;
  const int row0 = (slab / colSlabs) << 5;
  const int col0 = (slab % colSlabs) << 6;
  const int bz = blockIdx.z;
  const bf16_t* Ab = A + (long)bz * aBatch;
  const bf16_t* Wb = W + (long)bz * wBatch;

  v8f acc[2][4] = {};
  const int nk = Kdim >> 5;
  // double-buffered A fragments (software pipeline to hide load latency)
  v16bf a0 = load_frag_A(Ab, lda, row0,      0, lane, M);
  v16bf a1 = load_frag_A(Ab, lda, row0 + 16, 0, lane, M);
  for (int kc = 0; kc < nk; ++kc) {
    v16bf ca0 = a0, ca1 = a1;
    if (kc + 1 < nk) {
      a0 = load_frag_A(Ab, lda, row0,      kc + 1, lane, M);
      a1 = load_frag_A(Ab, lda, row0 + 16, kc + 1, lane, M);
    }
#pragma unroll
    for (int j = 0; j < 4; ++j) {
      v16bf b = load_frag_B(Wb, Kdim, col0 + j * 16, kc, lane);
      acc[0][j] = wmma_bf16(ca0, b, acc[0][j]);
      acc[1][j] = wmma_bf16(ca1, b, acc[1][j]);
    }
  }
  const int hr  = (lane & 16) ? 8 : 0;
  const int nlo = lane & 15;
#pragma unroll
  for (int rt = 0; rt < 2; ++rt) {
#pragma unroll
    for (int j = 0; j < 4; ++j) {
      int n = col0 + j * 16 + nlo;
      float bv = bias ? bias[(long)bz * biasBatch + n] : 0.f;
#pragma unroll
      for (int v = 0; v < 8; ++v) {
        int m = row0 + rt * 16 + hr + v;
        if (m < M) {
          float val = acc[rt][j][v] + bv;
          if (RELU) val = fmaxf(val, 0.f);
          long off = (long)bz * cBatch + (long)m * Ncol + n;
          if (OUTBF) ((bf16_t*)Cout)[off] = f2bf(val);
          else       ((float*)Cout)[off]  = val;
        }
      }
    }
  }
}

// ---------------- small utility kernels ----------------
__global__ void k_fill_f32(float* __restrict__ p, float v, long n) {
  long i = (long)blockIdx.x * 256 + threadIdx.x;
  if (i < n) p[i] = v;
}
__global__ void k_f32_to_bf16(const float* __restrict__ x, bf16_t* __restrict__ y, long n) {
  long i = (long)blockIdx.x * 256 + threadIdx.x;
  if (i < n) y[i] = f2bf(x[i]);
}
__global__ void k_init_h(const float* __restrict__ f, float* __restrict__ h,
                         bf16_t* __restrict__ hb) {
  int i = blockIdx.x * 256 + threadIdx.x;
  if (i < kN * kD) { float v = f[i]; h[i] = v; hb[i] = f2bf(v); }
}
// (O,I,K) f32 -> (O, K*I) bf16 so the GEMM K axis matches node-major im2col
__global__ void k_repack_conv(const float* __restrict__ w, bf16_t* __restrict__ y,
                              int O, int I, int K) {
  long idx = (long)blockIdx.x * 256 + threadIdx.x;
  long total = (long)O * I * K;
  if (idx >= total) return;
  int kk = (int)(idx % K);
  int i  = (int)((idx / K) % I);
  int o  = (int)(idx / ((long)K * I));
  y[(long)o * I * K + (long)kk * I + i] = f2bf(w[idx]);
}

// ---------------- GGNN edge gather/scatter ----------------
__global__ void k_gather_scatter(const bf16_t* __restrict__ Hall, const int* __restrict__ src,
                                 const int* __restrict__ dst, const int* __restrict__ et,
                                 float* __restrict__ a) {
  int e = blockIdx.x * 8 + (threadIdx.x >> 5);
  int lane = threadIdx.x & 31;
  if (e >= kE) return;
  // prefetch a far-ahead edge's Hall row into L2 (global_prefetch_b8) to hide
  // random-gather latency; one row = 256B = 2 cachelines.
  int ep = e + 4096;
  if (ep < kE && lane == 0) {
    const bf16_t* pm = Hall + ((long)et[ep] * kN + src[ep]) * kD;
    __builtin_prefetch(pm, 0, 1);
    __builtin_prefetch(pm + 64, 0, 1);
  }
  const bf16_t* m = Hall + ((long)et[e] * kN + src[e]) * kD + lane * 4;
  float* ad = a + (long)dst[e] * kD + lane * 4;
  atomicAdd(ad + 0, (float)m[0]);
  atomicAdd(ad + 1, (float)m[1]);
  atomicAdd(ad + 2, (float)m[2]);
  atomicAdd(ad + 3, (float)m[3]);
}

__global__ void k_gru(const float* __restrict__ gi, const float* __restrict__ gh,
                      float* __restrict__ h, bf16_t* __restrict__ hb) {
  int idx = blockIdx.x * 256 + threadIdx.x;
  if (idx >= kN * kD) return;
  int n = idx >> 7, d = idx & 127;
  const float* gin = gi + (long)n * 384;
  const float* ghn = gh + (long)n * 384;
  float r = 1.f / (1.f + expf(-(gin[d] + ghn[d])));
  float z = 1.f / (1.f + expf(-(gin[128 + d] + ghn[128 + d])));
  float g = tanhf(gin[256 + d] + r * ghn[256 + d]);
  float hn = (1.f - z) * g + z * h[idx];
  h[idx] = hn;
  hb[idx] = f2bf(hn);
}

// ---------------- GAT kernels ----------------
__global__ void k_attn_lr(const bf16_t* __restrict__ zf, const float* __restrict__ al,
                          const float* __restrict__ ar, float* __restrict__ el,
                          float* __restrict__ er) {
  int idx = blockIdx.x * 256 + threadIdx.x;
  if (idx >= kN * kH) return;
  int n = idx / kH, h = idx % kH;
  const bf16_t* z = zf + (long)n * (kH * kD) + h * kD;
  const float* wl = al + h * kD;
  const float* wr = ar + h * kD;
  float sl = 0.f, sr = 0.f;
  for (int d = 0; d < kD; ++d) { float zv = (float)z[d]; sl += zv * wl[d]; sr += zv * wr[d]; }
  el[idx] = sl; er[idx] = sr;
}

__global__ void k_edge_logit(const float* __restrict__ el, const float* __restrict__ er,
                             const int* __restrict__ src, const int* __restrict__ dst,
                             float* __restrict__ eeb, float* __restrict__ emax) {
  int idx = blockIdx.x * 256 + threadIdx.x;
  if (idx >= kE * kH) return;
  int e = idx / kH, h = idx % kH;
  float v = el[src[e] * kH + h] + er[dst[e] * kH + h];
  v = v > 0.f ? v : kSLOPE * v;
  eeb[idx] = v;
  float* addr = emax + dst[e] * kH + h;        // float atomic-max via int ordering
  if (v >= 0.f) atomicMax((int*)addr, __float_as_int(v));
  else          atomicMin((unsigned int*)addr, __float_as_uint(v));
}

__global__ void k_edge_exp(float* __restrict__ eeb, const float* __restrict__ emax,
                           float* __restrict__ den, const int* __restrict__ dst) {
  int idx = blockIdx.x * 256 + threadIdx.x;
  if (idx >= kE * kH) return;
  int e = idx / kH, h = idx % kH;
  float w = expf(eeb[idx] - emax[dst[e] * kH + h]);
  eeb[idx] = w;
  atomicAdd(&den[dst[e] * kH + h], w);
}

__global__ void k_edge_agg(const float* __restrict__ eeb, const float* __restrict__ den,
                           const bf16_t* __restrict__ zf, const int* __restrict__ src,
                           const int* __restrict__ dst, float* __restrict__ gout) {
  int e = blockIdx.x;
  int h = threadIdx.x >> 5;
  int lane = threadIdx.x & 31;
  float alpha = eeb[e * kH + h] / den[dst[e] * kH + h];
  const bf16_t* z = zf + (long)src[e] * (kH * kD) + h * kD + lane * 4;
  float* o = gout + (long)dst[e] * (kH * kD) + h * kD + lane * 4;
  atomicAdd(o + 0, alpha * (float)z[0]);
  atomicAdd(o + 1, alpha * (float)z[1]);
  atomicAdd(o + 2, alpha * (float)z[2]);
  atomicAdd(o + 3, alpha * (float)z[3]);
}

__global__ void k_head_concat(const float* __restrict__ gout, const float* __restrict__ gb,
                              const float* __restrict__ feat, bf16_t* __restrict__ hib,
                              bf16_t* __restrict__ cib) {
  int idx = blockIdx.x * 256 + threadIdx.x;
  if (idx >= kN * kD) return;
  int n = idx >> 7, d = idx & 127;
  const float* o = gout + (long)n * 384;
  float hm = (o[d] + gb[d] + o[128 + d] + gb[128 + d] + o[256 + d] + gb[256 + d]) * (1.f / 3.f);
  hib[idx] = f2bf(hm);
  cib[(long)n * 256 + d]       = f2bf(hm);
  cib[(long)n * 256 + 128 + d] = f2bf(feat[idx]);
}

// ---------------- pooling (node-major B x L x C) ----------------
template <bool MAXP, bool OUTF32>
__global__ void k_pool(const bf16_t* __restrict__ x, void* __restrict__ y,
                       int Lin, int C, int Lout, int kk, int ss) {
  long idx = (long)blockIdx.x * 256 + threadIdx.x;
  long total = (long)kB * Lout * C;
  if (idx >= total) return;
  int c = (int)(idx % C);
  int l = (int)((idx / C) % Lout);
  int b = (int)(idx / ((long)C * Lout));
  const bf16_t* xp = x + ((long)b * Lin + (long)l * ss) * C + c;
  float acc = MAXP ? -3.0e38f : 0.f;
  for (int i = 0; i < kk; ++i) {
    float v = (float)xp[(long)i * C];
    if (MAXP) acc = fmaxf(acc, v); else acc += v;
  }
  if (!MAXP) acc /= (float)kk;
  if (OUTF32) ((float*)y)[idx] = acc; else ((bf16_t*)y)[idx] = f2bf(acc);
}

// ---------------- final MLP pair + reduction ----------------
__global__ void k_finpair(const float* __restrict__ Y, const float* __restrict__ Z,
                          int CY, int CZ, int L,
                          const float* __restrict__ wy, const float* __restrict__ by,
                          const float* __restrict__ wz, const float* __restrict__ bz,
                          float* __restrict__ out) {
  __shared__ float red[256];
  int b = blockIdx.x, t = threadIdx.x;
  float acc = 0.f;
  for (int l = t; l < L; l += 256) {
    const float* yr = Y + ((long)b * L + l) * CY;
    float s1 = by[0];
    for (int i = 0; i < CY; ++i) s1 += yr[i] * wy[i];
    const float* zr = Z + ((long)b * L + l) * CZ;
    float s2 = bz[0];
    for (int i = 0; i < CZ; ++i) s2 += zr[i] * wz[i];
    acc += s1 * s2;
  }
  red[t] = acc; __syncthreads();
  for (int o = 128; o > 0; o >>= 1) { if (t < o) red[t] += red[t + o]; __syncthreads(); }
  if (t == 0) out[b] = red[0] / (float)L;
}

__global__ void k_sigout(const float* __restrict__ ba, float* __restrict__ out) {
  int b = threadIdx.x;
  if (b < kB) out[b] = 1.f / (1.f + expf(-0.5f * (ba[b] + ba[16 + b])));
}

// ---------------- host-side GEMM dispatcher ----------------
static void launch_gemm(const bf16_t* A, int lda, long aB, const bf16_t* W, long wB,
                        const float* bias, long bB, void* C, long cB,
                        int M, int Ncol, int K, int nz, bool outbf, bool relu,
                        hipStream_t s) {
  int slabs = ((M + 31) / 32) * (Ncol / 64);
  dim3 grid((slabs + 3) / 4, 1, nz);
  dim3 block(32, 4, 1);
  if (outbf) {
    if (relu) k_wmma_gemm<true, true ><<<grid, block, 0, s>>>(A, lda, aB, W, wB, bias, bB, C, cB, M, Ncol, K);
    else      k_wmma_gemm<true, false><<<grid, block, 0, s>>>(A, lda, aB, W, wB, bias, bB, C, cB, M, Ncol, K);
  } else {
    if (relu) k_wmma_gemm<false, true ><<<grid, block, 0, s>>>(A, lda, aB, W, wB, bias, bB, C, cB, M, Ncol, K);
    else      k_wmma_gemm<false, false><<<grid, block, 0, s>>>(A, lda, aB, W, wB, bias, bB, C, cB, M, Ncol, K);
  }
}

// ---------------- entry point ----------------
extern "C" void kernel_launch(void* const* d_in, const int* in_sizes, int n_in,
                              void* d_out, int out_size, void* d_ws, size_t ws_size,
                              hipStream_t stream) {
  (void)in_sizes; (void)n_in; (void)out_size; (void)ws_size;
  const float* features = (const float*)d_in[0];
  const int*   src      = (const int*)d_in[1];
  const int*   dst      = (const int*)d_in[2];
  const int*   etype    = (const int*)d_in[3];
  const float* W_et     = (const float*)d_in[4];
  const float* b_et     = (const float*)d_in[5];
  const float* W_ih     = (const float*)d_in[6];
  const float* b_ih     = (const float*)d_in[7];
  const float* W_hh     = (const float*)d_in[8];
  const float* b_hh     = (const float*)d_in[9];
  const float* Wg       = (const float*)d_in[10];
  const float* attn_l   = (const float*)d_in[11];
  const float* attn_r   = (const float*)d_in[12];
  const float* gat_bias = (const float*)d_in[13];
  const float* conv1_w  = (const float*)d_in[14];
  const float* conv1_b  = (const float*)d_in[15];
  const float* avg1_w   = (const float*)d_in[16];
  const float* avg1_b   = (const float*)d_in[17];
  const float* conv2_w  = (const float*)d_in[18];
  const float* conv2_b  = (const float*)d_in[19];
  const float* avg2_w   = (const float*)d_in[20];
  const float* avg2_b   = (const float*)d_in[21];
  const float* cc1_w    = (const float*)d_in[22];
  const float* cc1_b    = (const float*)d_in[23];
  const float* acc1_w   = (const float*)d_in[24];
  const float* acc1_b   = (const float*)d_in[25];
  const float* cc2_w    = (const float*)d_in[26];
  const float* cc2_b    = (const float*)d_in[27];
  const float* acc2_w   = (const float*)d_in[28];
  const float* acc2_b   = (const float*)d_in[29];
  const float* mlph_w   = (const float*)d_in[30];
  const float* mlph_b   = (const float*)d_in[31];
  const float* mlp_w    = (const float*)d_in[32];
  const float* mlp_b    = (const float*)d_in[33];

  char* ws = (char*)d_ws;
  float*  Hbuf = (float*)(ws + OFF_H);
  bf16_t* HB   = (bf16_t*)(ws + OFF_HB);
  bf16_t* HALL = (bf16_t*)(ws + OFF_HALL);
  float*  Aagg = (float*)(ws + OFF_A);
  bf16_t* ABF  = (bf16_t*)(ws + OFF_ABF);
  float*  GI   = (float*)(ws + OFF_GI);
  float*  GH   = (float*)(ws + OFF_GH);
  bf16_t* ZF   = (bf16_t*)(ws + OFF_ZF);
  float*  EL   = (float*)(ws + OFF_EL);
  float*  ER   = (float*)(ws + OFF_ER);
  float*  EMAX = (float*)(ws + OFF_EMAX);
  float*  DEN  = (float*)(ws + OFF_DEN);
  float*  EEb  = (float*)(ws + OFF_EE);
  float*  GOUT = (float*)(ws + OFF_GOUT);
  bf16_t* HIB  = (bf16_t*)(ws + OFF_HIB);
  bf16_t* CIB  = (bf16_t*)(ws + OFF_CIB);
  bf16_t* BUFA = (bf16_t*)(ws + OFF_BUFA);
  bf16_t* BUFB = (bf16_t*)(ws + OFF_BUFB);
  float*  Y2   = (float*)(ws + OFF_Y2);
  float*  P2   = (float*)(ws + OFF_P2);
  float*  Z2v  = (float*)(ws + OFF_Z2);
  float*  U2   = (float*)(ws + OFF_U2);
  float*  BA   = (float*)(ws + OFF_BA);
  bf16_t* WB   = (bf16_t*)(ws + OFF_WB);

  auto cvt = [&](const float* sp, bf16_t* dp, long n) {
    k_f32_to_bf16<<<(int)((n + 255) / 256), 256, 0, stream>>>(sp, dp, n);
  };
  auto rpk = [&](const float* w, bf16_t* y, int O, int I, int K) {
    long n = (long)O * I * K;
    k_repack_conv<<<(int)((n + 255) / 256), 256, 0, stream>>>(w, y, O, I, K);
  };
  // weight prep (bf16)
  cvt(W_et, WB + WB_WET, (long)kNET * kD * kD);
  cvt(W_ih, WB + WB_WIH, 384 * 128);
  cvt(W_hh, WB + WB_WHH, 384 * 128);
  cvt(Wg,   WB + WB_WG,  384 * 128);
  rpk(conv1_w, WB + WB_C1, 128, 128, 3);
  rpk(avg1_w,  WB + WB_A1, 128, 128, 3);
  cvt(conv2_w, WB + WB_C2, 128 * 128);
  cvt(avg2_w,  WB + WB_A2, 128 * 128);
  rpk(cc1_w,  WB + WB_CC1, 256, 256, 3);
  rpk(acc1_w, WB + WB_AC1, 256, 256, 3);
  cvt(cc2_w,  WB + WB_CC2, 256 * 256);
  cvt(acc2_w, WB + WB_AC2, 256 * 256);

  const int ND = kN * kD;
  k_init_h<<<(ND + 255) / 256, 256, 0, stream>>>(features, Hbuf, HB);

  // ---- GatedGraphConv: 8 steps ----
  for (int s = 0; s < kSTEPS; ++s) {
    k_fill_f32<<<(ND + 255) / 256, 256, 0, stream>>>(Aagg, 0.f, (long)ND);
    // Hall[t] = h @ W_et[t]^T + b_et[t]  (bf16 out, batched over edge type)
    launch_gemm(HB, 128, 0, WB + WB_WET, (long)kD * kD, b_et, kD,
                HALL, (long)kN * kD, kN, kD, kD, kNET, true, false, stream);
    k_gather_scatter<<<kE / 8, 256, 0, stream>>>(HALL, src, dst, etype, Aagg);
    k_f32_to_bf16<<<(ND + 255) / 256, 256, 0, stream>>>(Aagg, ABF, (long)ND);
    launch_gemm(ABF, 128, 0, WB + WB_WIH, 0, b_ih, 0, GI, 0, kN, 384, 128, 1, false, false, stream);
    launch_gemm(HB,  128, 0, WB + WB_WHH, 0, b_hh, 0, GH, 0, kN, 384, 128, 1, false, false, stream);
    k_gru<<<(ND + 255) / 256, 256, 0, stream>>>(GI, GH, Hbuf, HB);
  }

  // ---- GATConv ----
  launch_gemm(HB, 128, 0, WB + WB_WG, 0, nullptr, 0, ZF, 0, kN, 384, 128, 1, true, false, stream);
  k_attn_lr<<<(kN * kH + 255) / 256, 256, 0, stream>>>(ZF, attn_l, attn_r, EL, ER);
  k_fill_f32<<<(kN * kH + 255) / 256, 256, 0, stream>>>(EMAX, -3.0e38f, (long)kN * kH);
  k_fill_f32<<<(kN * kH + 255) / 256, 256, 0, stream>>>(DEN, 0.f, (long)kN * kH);
  k_fill_f32<<<((long)kN * 384 + 255) / 256, 256, 0, stream>>>(GOUT, 0.f, (long)kN * 384);
  k_edge_logit<<<(kE * kH + 255) / 256, 256, 0, stream>>>(EL, ER, src, dst, EEb, EMAX);
  k_edge_exp<<<(kE * kH + 255) / 256, 256, 0, stream>>>(EEb, EMAX, DEN, dst);
  k_edge_agg<<<kE, 96, 0, stream>>>(EEb, DEN, ZF, src, dst, GOUT);
  k_head_concat<<<(ND + 255) / 256, 256, 0, stream>>>(GOUT, gat_bias, features, HIB, CIB);

  // ---- conv/pool head (WMMA with implicit im2col; node-major B x L x C) ----
  const int L0 = 2048, L1 = 2046, L2 = 1022, L3 = 511;
  // max path on h
  launch_gemm(HIB, 128, (long)L0 * 128, WB + WB_C1, 0, conv1_b, 0,
              BUFA, (long)L1 * 128, L1, 128, 384, kB, true, true, stream);
  k_pool<true, false><<<((long)kB * L2 * 128 + 255) / 256, 256, 0, stream>>>(BUFA, BUFB, L1, 128, L2, 3, 2);
  launch_gemm(BUFB, 128, (long)L2 * 128, WB + WB_C2, 0, conv2_b, 0,
              BUFA, (long)L2 * 128, L2, 128, 128, kB, true, true, stream);
  k_pool<true, true><<<((long)kB * L3 * 128 + 255) / 256, 256, 0, stream>>>(BUFA, Y2, L2, 128, L3, 2, 2);
  // avg path on h
  launch_gemm(HIB, 128, (long)L0 * 128, WB + WB_A1, 0, avg1_b, 0,
              BUFA, (long)L1 * 128, L1, 128, 384, kB, true, true, stream);
  k_pool<false, false><<<((long)kB * L2 * 128 + 255) / 256, 256, 0, stream>>>(BUFA, BUFB, L1, 128, L2, 3, 2);
  launch_gemm(BUFB, 128, (long)L2 * 128, WB + WB_A2, 0, avg2_b, 0,
              BUFA, (long)L2 * 128, L2, 128, 128, kB, true, true, stream);
  k_pool<false, true><<<((long)kB * L3 * 128 + 255) / 256, 256, 0, stream>>>(BUFA, P2, L2, 128, L3, 2, 2);
  // max path on concat (C=256)
  launch_gemm(CIB, 256, (long)L0 * 256, WB + WB_CC1, 0, cc1_b, 0,
              BUFA, (long)L1 * 256, L1, 256, 768, kB, true, true, stream);
  k_pool<true, false><<<((long)kB * L2 * 256 + 255) / 256, 256, 0, stream>>>(BUFA, BUFB, L1, 256, L2, 3, 2);
  launch_gemm(BUFB, 256, (long)L2 * 256, WB + WB_CC2, 0, cc2_b, 0,
              BUFA, (long)L2 * 256, L2, 256, 256, kB, true, true, stream);
  k_pool<true, true><<<((long)kB * L3 * 256 + 255) / 256, 256, 0, stream>>>(BUFA, Z2v, L2, 256, L3, 2, 2);
  // avg path on concat
  launch_gemm(CIB, 256, (long)L0 * 256, WB + WB_AC1, 0, acc1_b, 0,
              BUFA, (long)L1 * 256, L1, 256, 768, kB, true, true, stream);
  k_pool<false, false><<<((long)kB * L2 * 256 + 255) / 256, 256, 0, stream>>>(BUFA, BUFB, L1, 256, L2, 3, 2);
  launch_gemm(BUFB, 256, (long)L2 * 256, WB + WB_AC2, 0, acc2_b, 0,
              BUFA, (long)L2 * 256, L2, 256, 256, kB, true, true, stream);
  k_pool<false, true><<<((long)kB * L3 * 256 + 255) / 256, 256, 0, stream>>>(BUFA, U2, L2, 256, L3, 2, 2);

  // ---- final MLP pair, mean, sigmoid ----
  k_finpair<<<kB, 256, 0, stream>>>(Y2, Z2v, 128, 256, L3, mlph_w, mlph_b, mlp_w, mlp_b, BA);
  k_finpair<<<kB, 256, 0, stream>>>(P2, U2, 128, 256, L3, mlph_w, mlph_b, mlp_w, mlp_b, BA + 16);
  k_sigout<<<1, 32, 0, stream>>>(BA, (float*)d_out);
}
